// PointerDecoder_62758062129674
// MI455X (gfx1250) — matrix operand order repaired and checked by reading, not compiled, not run on hardware
//
#include <hip/hip_runtime.h>
#include <hip/hip_bf16.h>
#include <cstdint>
#include <cstddef>

// Problem constants (from reference)
#define SEQ  512
#define BAT  256
#define DM   1024
#define NH   8
#define DH   128
#define NS   64
#define MPAD 16   // A-matrix rows (8 heads + 8 zero-padded)

typedef __attribute__((ext_vector_type(16))) __bf16 v16bf;
typedef __attribute__((ext_vector_type(8)))  float  v8f;

struct __align__(16) U128 { unsigned int x, y, z, w; };
struct __align__(32) U256 { U128 lo, hi; };
union AFrag { U256 q; v16bf v; };

__device__ inline uint16_t f2bf(float x) {
    uint32_t u = __float_as_uint(x);
    uint32_t r = u + 0x7FFFu + ((u >> 16) & 1u);   // round-to-nearest-even
    return (uint16_t)(r >> 16);
}
// cheap truncating pack for the HBM-bound streaming path (2-3 VALU, v_perm-able)
__device__ inline unsigned int packbf2_t(float a, float b) {
    return (__float_as_uint(a) >> 16) | (__float_as_uint(b) & 0xffff0000u);
}
__device__ inline uint32_t hash3(uint32_t a, uint32_t b, uint32_t c) {
    uint32_t h = a * 0x9E3779B1u ^ b * 0x85EBCA77u ^ c * 0xC2B2AE3Du ^ 0x27220A95u;
    h ^= h >> 16; h *= 0x7FEB352Du;
    h ^= h >> 15; h *= 0x846CA68Bu;
    h ^= h >> 16;
    return h;
}

// ---------------------------------------------------------------------------
// Kernel 1: qh[b, j] = (memory[0,b,:] . Wq[j,:] + bq[j]) * rsqrt(128)
// Tiled 32x32 fp32 GEMM. grid = (B/32, D/32), block = 256.
// ---------------------------------------------------------------------------
__global__ __launch_bounds__(256) void qh_kernel(
    const float* __restrict__ mem0,   // memory[0] = [B, D]
    const float* __restrict__ Wq,     // [D, D]
    const float* __restrict__ bq,     // [D]
    float* __restrict__ qh)           // [B, D]
{
    const int b0 = blockIdx.x * 32;
    const int j0 = blockIdx.y * 32;
    const int tx = threadIdx.x & 31;
    const int ty = threadIdx.x >> 5;          // 0..7
    __shared__ float As[32][33];
    __shared__ float Bs[32][33];
    float acc[4] = {0.f, 0.f, 0.f, 0.f};

    for (int k0 = 0; k0 < DM; k0 += 32) {
        #pragma unroll
        for (int r = 0; r < 4; ++r) {
            int row = ty + r * 8;
            As[row][tx] = mem0[(size_t)(b0 + row) * DM + k0 + tx];
            Bs[row][tx] = Wq[(size_t)(j0 + row) * DM + k0 + tx];
        }
        __syncthreads();
        #pragma unroll
        for (int k = 0; k < 32; ++k) {
            float bv = Bs[tx][k];
            #pragma unroll
            for (int r = 0; r < 4; ++r)
                acc[r] = fmaf(As[ty + r * 8][k], bv, acc[r]);
        }
        __syncthreads();
    }
    const float scale = 0.088388347648318447f;   // 1/sqrt(128)
    float bias = bq[j0 + tx];
    #pragma unroll
    for (int r = 0; r < 4; ++r)
        qh[(size_t)(b0 + ty + r * 8) * DM + j0 + tx] = (acc[r] + bias) * scale;
}

// ---------------------------------------------------------------------------
// Kernel 2: u[b,h,d] = qh[b,h,:] . Wk[h*128 + :, d]  (bf16, zero-padded to 16
//           rows per b so the WMMA A-fragment load needs no lane predication)
//           c[b,h]  = qh[b,h,:] . bk[h*128 + :]
// grid = B*16, block = 256.
// ---------------------------------------------------------------------------
__global__ __launch_bounds__(256) void u_kernel(
    const float* __restrict__ qh,
    const float* __restrict__ Wk,
    const float* __restrict__ bk,
    uint16_t* __restrict__ ubf,       // [B*MPAD, D] bf16 (rows 8..15 zero)
    float* __restrict__ cb)           // [B*H]
{
    const int blk = blockIdx.x;
    const int b = blk >> 4, h = blk & 15;
    const int tid = threadIdx.x;

    if (h >= NH) {                    // zero padding rows
        #pragma unroll
        for (int rep = 0; rep < 4; ++rep)
            ubf[(size_t)blk * DM + tid + rep * 256] = 0;
        return;
    }

    __shared__ float qs[DH];
    if (tid < DH) qs[tid] = qh[(size_t)b * DM + h * DH + tid];
    __syncthreads();

    const float* wkB = Wk + (size_t)h * DH * DM;
    #pragma unroll 1
    for (int rep = 0; rep < 4; ++rep) {
        int d = tid + rep * 256;
        float acc = 0.f;
        #pragma unroll 4
        for (int j = 0; j < DH; ++j)
            acc = fmaf(qs[j], wkB[(size_t)j * DM + d], acc);
        ubf[(size_t)blk * DM + d] = f2bf(acc);
    }
    if (tid == 0) {
        float c = 0.f;
        const float* bkh = bk + h * DH;
        for (int j = 0; j < DH; ++j) c = fmaf(qs[j], bkh[j], c);
        cb[b * NH + h] = c;
    }
}

// ---------------------------------------------------------------------------
// Kernel 3 (HBM-bound, WMMA): logits[b,h,s] = sum_d u_bf16[b,h,d]*mem_bf16[s,b,d] + c[b,h]
// Register-direct streaming: B fragments loaded straight from global
// (each wave covers 16 rows x 128B of fully-consumed cachelines per tile),
// truncating fp32->bf16 pack, no LDS, no barriers, unconditional A load
// -> compiler pipelines loads across K-chunks.
// grid = (B, 2): 512 blocks, 8 waves, 2 s-tiles per wave.
// ---------------------------------------------------------------------------
__global__ __launch_bounds__(256) void logits_kernel(
    const float* __restrict__ mem,    // [S, B, D]
    const uint16_t* __restrict__ ubf, // [B*MPAD, D]
    const float* __restrict__ cb,     // [B*H]
    float* __restrict__ lg)           // [B*H, S]
{
    const int b     = blockIdx.x;
    const int sBase = blockIdx.y * 256;
    const int tid   = threadIdx.x;
    const int lane  = tid & 31;
    const int wv    = tid >> 5;              // 0..7
    const int m     = lane & 15;
    const bool hi   = lane >= 16;

    v8f acc[2];
    #pragma unroll
    for (int t = 0; t < 2; ++t) acc[t] = (v8f){0.f,0.f,0.f,0.f,0.f,0.f,0.f,0.f};

    // Per-lane base pointers (chunk-invariant)
    const int koff = hi ? 16 : 0;            // A-matrix K half per lane (contiguous, matches B)
    const int boff = hi ? 16 : 0;            // B-matrix K half per lane
    const uint16_t* uBase = ubf + (size_t)(b * MPAD + m) * DM + koff;  // rows >=8 are zeros
    const float* bBase[2];
    #pragma unroll
    for (int t = 0; t < 2; ++t) {
        int s = sBase + (wv * 2 + t) * 16 + m;      // B column = sequence position
        bBase[t] = mem + (size_t)s * (BAT * DM) + (size_t)b * DM + boff;
    }

    for (int c = 0; c < 32; ++c) {
        const int d0 = c * 32;

        // A fragment: rows = heads (8 valid, 8 zero-padded) -- unconditional
        AFrag af;
        af.q.lo = *(const U128*)(uBase + d0);
        af.q.hi = *(const U128*)(uBase + d0 + 8);

        // B fragments: 16 contiguous fp32 per lane -> 16 bf16 (K-half per lane)
        AFrag bfr[2];
        #pragma unroll
        for (int t = 0; t < 2; ++t) {
            const float* p = bBase[t] + d0;
            const float4 a0 = *(const float4*)(p);
            const float4 a1 = *(const float4*)(p + 4);
            const float4 a2 = *(const float4*)(p + 8);
            const float4 a3 = *(const float4*)(p + 12);
            bfr[t].q.lo = U128{ packbf2_t(a0.x, a0.y), packbf2_t(a0.z, a0.w),
                                packbf2_t(a1.x, a1.y), packbf2_t(a1.z, a1.w) };
            bfr[t].q.hi = U128{ packbf2_t(a2.x, a2.y), packbf2_t(a2.z, a2.w),
                                packbf2_t(a3.x, a3.y), packbf2_t(a3.z, a3.w) };
        }

        #pragma unroll
        for (int t = 0; t < 2; ++t)
            acc[t] = __builtin_amdgcn_wmma_f32_16x16x32_bf16(
                false, af.v, false, bfr[t].v, (short)0, acc[t], false, false);
    }

    // C layout: lanes 0..15, VGPR r -> (head=r, s=tile*16+lane)
    if (!hi) {
        #pragma unroll
        for (int t = 0; t < 2; ++t) {
            int s = sBase + (wv * 2 + t) * 16 + m;
            #pragma unroll
            for (int r = 0; r < NH; ++r)
                lg[((size_t)b * NH + r) * SEQ + s] = acc[t][r] + cb[b * NH + r];
        }
    }
}

// ---------------------------------------------------------------------------
// Kernel 4: 64-step masked-softmax pointer decode, one block per batch b.
// E[h,s]=exp(l-max) precomputed; running denominators updated O(H) per step;
// categorical sampling via Gumbel-max argmax reduction.
// ---------------------------------------------------------------------------
__global__ __launch_bounds__(512) void decode_kernel(
    const float* __restrict__ lg,     // [B*H, S]
    float* __restrict__ out)          // [B, NS]
{
    const int b    = blockIdx.x;
    const int tid  = threadIdx.x;     // == s
    const int lane = tid & 31;
    const int wv   = tid >> 5;        // 0..15

    __shared__ float E[NH * SEQ];     // 16 KB
    __shared__ float Sh[NH], invS[NH];
    __shared__ unsigned char msk[SEQ];
    __shared__ float redf[16];
    __shared__ int   redi[16];
    __shared__ float bcf;

    float l[NH];
    #pragma unroll
    for (int h = 0; h < NH; ++h)
        l[h] = lg[((size_t)b * NH + h) * SEQ + tid];
    msk[tid] = (tid == 0) ? 1 : 0;
    __syncthreads();

    // init: per-head max, E, and running denominator excluding s==0
    for (int h = 0; h < NH; ++h) {
        float v = l[h];
        #pragma unroll
        for (int o = 16; o > 0; o >>= 1) v = fmaxf(v, __shfl_xor(v, o, 32));
        if (lane == 0) redf[wv] = v;
        __syncthreads();
        if (tid == 0) {
            float mx = redf[0];
            for (int i = 1; i < 16; ++i) mx = fmaxf(mx, redf[i]);
            bcf = mx;
        }
        __syncthreads();
        float e = __expf(l[h] - bcf);
        E[h * SEQ + tid] = e;
        float se = (tid == 0) ? 0.f : e;
        #pragma unroll
        for (int o = 16; o > 0; o >>= 1) se += __shfl_xor(se, o, 32);
        if (lane == 0) redf[wv] = se;
        __syncthreads();
        if (tid == 0) {
            float ssum = 0.f;
            for (int i = 0; i < 16; ++i) ssum += redf[i];
            Sh[h] = ssum;
            invS[h] = 1.0f / ssum;
        }
        __syncthreads();
    }

    const float NEG_BIG = -3.402823466e38f;   // finfo(float32).min
    for (int step = 0; step < NS; ++step) {
        const bool mk = msk[tid] != 0;
        float w = 0.f;
        #pragma unroll
        for (int h = 0; h < NH; ++h) w = fmaf(E[h * SEQ + tid], invS[h], w);
        w *= 0.125f;
        float score = mk ? NEG_BIG : w;

        uint32_t hsh = hash3((uint32_t)b, (uint32_t)tid, (uint32_t)step);
        float ur  = (float)(hsh >> 8) * (1.0f / 16777216.0f) + 1e-12f;  // (0,1)
        float key = score - __logf(-__logf(ur));                        // Gumbel-max

        float kv = key; int ki = tid;
        #pragma unroll
        for (int o = 16; o > 0; o >>= 1) {
            float ov = __shfl_xor(kv, o, 32);
            int   oi = __shfl_xor(ki, o, 32);
            if (ov > kv || (ov == kv && oi < ki)) { kv = ov; ki = oi; }
        }
        if (lane == 0) { redf[wv] = kv; redi[wv] = ki; }
        __syncthreads();
        if (tid == 0) {
            float bv = redf[0]; int bi = redi[0];
            for (int i = 1; i < 16; ++i)
                if (redf[i] > bv || (redf[i] == bv && redi[i] < bi)) { bv = redf[i]; bi = redi[i]; }
            out[(size_t)b * NS + step] = (float)bi;   // gen.T layout [B, NS]
            msk[bi] = 1;
            #pragma unroll
            for (int h = 0; h < NH; ++h) {
                float ns = Sh[h] - E[h * SEQ + bi];
                Sh[h] = ns;
                invS[h] = 1.0f / ns;
            }
        }
        __syncthreads();
    }
}

// ---------------------------------------------------------------------------
extern "C" void kernel_launch(void* const* d_in, const int* in_sizes, int n_in,
                              void* d_out, int out_size, void* d_ws, size_t ws_size,
                              hipStream_t stream)
{
    const float* memory = (const float*)d_in[0];   // [S, B, D]
    // d_in[1] = tgt (unused by reference math)
    const float* Wq = (const float*)d_in[2];
    const float* bq = (const float*)d_in[3];
    const float* Wk = (const float*)d_in[4];
    const float* bk = (const float*)d_in[5];
    // d_in[6] = num_steps (compile-time 64)

    char* ws = (char*)d_ws;
    float*    qh  = (float*)ws;                               // 1 MB
    uint16_t* ubf = (uint16_t*)(ws + (1u << 20));             // 8 MB (16 rows/b)
    float*    cb  = (float*)(ws + (9u << 20));                // 8 KB
    float*    lg  = (float*)(ws + (9u << 20) + (1u << 16));   // 4 MB
    float*    out = (float*)d_out;

    qh_kernel<<<dim3(BAT / 32, DM / 32), 256, 0, stream>>>(memory, Wq, bq, qh);
    u_kernel<<<dim3(BAT * MPAD), 256, 0, stream>>>(qh, Wk, bk, ubf, cb);
    logits_kernel<<<dim3(BAT, 2), 256, 0, stream>>>(memory, ubf, cb, lg);
    decode_kernel<<<dim3(BAT), 512, 0, stream>>>(lg, out);
}
